// FuzzyNeuron_32693291057854
// MI455X (gfx1250) — matrix operand order repaired and checked by reading, not compiled, not run on hardware
//
#include <hip/hip_runtime.h>

typedef __attribute__((ext_vector_type(16))) _Float16 v16h;
typedef __attribute__((ext_vector_type(8)))  float    v8f;

#define A_DIM 16
#define R_DIM 64
#define NCOL  128      // 64 z columns | 64 logw columns
#define NTILE 8        // 128 / 16
#define K_DIM 32       // [x | x^2]

// d_ws layout:
//   _Float16 Bpack[2][NTILE][32][16]   (hi/lo, tile, lane, slot)  = 16 KB
//   float    bias[128]                 (bias_z[64], bias_w[64])   = 512 B

// ---------------------------------------------------------------------------
// Prep: build B (32x128) from mu/sigma/rho, split hi/lo, pre-swizzle into the
// per-lane WMMA B-operand layout, and compute the per-rule biases.
//   B[K][r]      = rho[r][K]            (K<16)   z columns
//   B[K][64+r]   = mu/sigma^2           (K<16)   logw: coefficient of x
//   B[16+a][64+r]= -1/(2 sigma^2)                logw: coefficient of x^2
//   bias_z[r] = rho[r][16] ;  bias_w[r] = -sum_a mu^2/(2 sigma^2)
// ---------------------------------------------------------------------------
__global__ void fuzzy_prep_kernel(const float* __restrict__ mu,
                                  const float* __restrict__ sigma,
                                  const float* __restrict__ rho,
                                  _Float16* __restrict__ Bpack,
                                  float* __restrict__ bias) {
  __shared__ float Bf[K_DIM][NCOL];
  const int tid = threadIdx.x;

  for (int idx = tid; idx < K_DIM * NCOL; idx += blockDim.x) {
    const int K = idx / NCOL, col = idx % NCOL;
    float v;
    if (col < R_DIM) {                       // z columns
      v = (K < A_DIM) ? rho[col * (A_DIM + 1) + K] : 0.0f;
    } else {                                 // logw columns
      const int r = col - R_DIM;
      const int a = K & 15;
      const float s   = sigma[r * A_DIM + a];
      const float inv = 1.0f / (s * s);
      v = (K < A_DIM) ? mu[r * A_DIM + a] * inv : -0.5f * inv;
    }
    Bf[K][col] = v;
  }
  __syncthreads();

  // Pack hi/lo into ISA B-operand layout: lane holds column n = lane%16,
  // slot s -> K=8h+s (s<8), K=16+8h+(s-8) (s>=8).
  for (int p = tid; p < 2 * NTILE * 32 * 16; p += blockDim.x) {
    const int hl   = p >> 12;
    const int rem  = p & 4095;
    const int t    = rem >> 9;
    const int lane = (rem >> 4) & 31;
    const int s    = rem & 15;
    const int K    = (s < 8) ? 8 * (lane >> 4) + s : 16 + 8 * (lane >> 4) + (s - 8);
    const int col  = t * 16 + (lane & 15);
    const float v  = Bf[K][col];
    const _Float16 hi = (_Float16)v;
    Bpack[p] = hl ? (_Float16)(v - (float)hi) : hi;
  }

  if (tid < R_DIM) {
    bias[tid] = rho[tid * (A_DIM + 1) + A_DIM];        // bias_z
    float c0 = 0.0f;
    for (int a = 0; a < A_DIM; ++a) {
      const float s = sigma[tid * A_DIM + a];
      const float m = mu[tid * A_DIM + a];
      c0 -= 0.5f * m * m / (s * s);
    }
    bias[R_DIM + tid] = c0;                            // bias_w
  }
}

// ---------------------------------------------------------------------------
// Main: each wave owns a 16-row block. A = [x | x^2] as 16x32 f16 (hi/lo),
// 8 output tiles of 16x16, 3 WMMAs each (hi*hi + lo*hi + hi*lo).
// Epilogue: w = exp(logw + bias_w), num/den row-sums over r, lane reduction.
// ---------------------------------------------------------------------------
__global__ void __launch_bounds__(256)
fuzzy_main_kernel(const float* __restrict__ x,
                  const _Float16* __restrict__ Bpack,
                  const float* __restrict__ biasg,
                  float* __restrict__ out,
                  int row_tiles, int total_waves) {
  __shared__ __align__(32) _Float16 Bsh[2 * NTILE * 32 * 16];
  __shared__ float biassh[2 * R_DIM];

  const int tid = threadIdx.x;
  // stage B (16 KB) + biases into LDS once per block
  {
    const uint4* src = (const uint4*)Bpack;
    uint4*       dst = (uint4*)Bsh;
    for (int i = tid; i < (2 * NTILE * 32 * 16) / 8; i += blockDim.x) dst[i] = src[i];
    for (int i = tid; i < 2 * R_DIM; i += blockDim.x) biassh[i] = biasg[i];
  }
  __syncthreads();

  const int lane  = tid & 31;
  const int h     = lane >> 4;        // lane half
  const int colr  = lane & 15;        // matrix row (A) / column (B,C)
  const int wave  = (blockIdx.x * blockDim.x + tid) >> 5;

  for (int tile = wave; tile < row_tiles; tile += total_waves) {
    // ---- load this lane's 8 row elements and build A_hi / A_lo -----------
    const float* xp = x + ((size_t)tile * 16 + colr) * A_DIM + h * 8;
    __builtin_prefetch(xp + (size_t)total_waves * 16 * A_DIM, 0, 0);
    const float4 x0 = ((const float4*)xp)[0];
    const float4 x1 = ((const float4*)xp)[1];
    const float xv[8] = {x0.x, x0.y, x0.z, x0.w, x1.x, x1.y, x1.z, x1.w};

    v16h a_hi, a_lo;
#pragma unroll
    for (int s = 0; s < 8; ++s) {
      const float v = xv[s];
      const _Float16 vh = (_Float16)v;
      a_hi[s] = vh;                                  // K slots 0..7  : x
      a_lo[s] = (_Float16)(v - (float)vh);
      const float q = v * v;
      const _Float16 qh = (_Float16)q;
      a_hi[s + 8] = qh;                              // K slots 8..15 : x^2
      a_lo[s + 8] = (_Float16)(q - (float)qh);
    }

    // ---- 8 tiles x 3 chained WMMAs ---------------------------------------
    v8f c[NTILE];
#pragma unroll
    for (int t = 0; t < NTILE; ++t)
#pragma unroll
      for (int i = 0; i < 8; ++i) c[t][i] = 0.0f;

#pragma unroll
    for (int t = 0; t < NTILE; ++t) {
      const v16h bh = *(const v16h*)&Bsh[(0 * NTILE + t) * 512 + lane * 16];
      const v16h bl = *(const v16h*)&Bsh[(1 * NTILE + t) * 512 + lane * 16];
      c[t] = __builtin_amdgcn_wmma_f32_16x16x32_f16(false, a_hi, false, bh,
                                                    (short)0, c[t], false, false);
      c[t] = __builtin_amdgcn_wmma_f32_16x16x32_f16(false, a_lo, false, bh,
                                                    (short)0, c[t], false, false);
      c[t] = __builtin_amdgcn_wmma_f32_16x16x32_f16(false, a_hi, false, bl,
                                                    (short)0, c[t], false, false);
    }

    // ---- epilogue: num/den partial sums over this lane's columns ---------
    float num[8], den[8];
#pragma unroll
    for (int i = 0; i < 8; ++i) { num[i] = 0.0f; den[i] = 0.0f; }

#pragma unroll
    for (int t = 0; t < 4; ++t) {                    // tiles t: z, t+4: logw
      const float bz = biassh[t * 16 + colr];
      const float bw = biassh[R_DIM + t * 16 + colr];
#pragma unroll
      for (int i = 0; i < 8; ++i) {
        const float z = c[t][i] + bz;
        const float w = __expf(c[t + 4][i] + bw);
        num[i] = fmaf(z, w, num[i]);
        den[i] += w;
      }
    }

    // reduce across the 16 lanes that share each row group
#pragma unroll
    for (int i = 0; i < 8; ++i) {
#pragma unroll
      for (int m = 1; m < 16; m <<= 1) {
        num[i] += __shfl_xor(num[i], m, 32);
        den[i] += __shfl_xor(den[i], m, 32);
      }
    }

    if (colr == 0) {
      const int rowbase = tile * 16 + h * 8;
#pragma unroll
      for (int i = 0; i < 8; ++i)
        out[rowbase + i] = num[i] / (den[i] + 1e-13f);
    }
  }
}

// ---------------------------------------------------------------------------
extern "C" void kernel_launch(void* const* d_in, const int* in_sizes, int n_in,
                              void* d_out, int out_size, void* d_ws, size_t ws_size,
                              hipStream_t stream) {
  const float* x     = (const float*)d_in[0];
  const float* mu    = (const float*)d_in[1];
  const float* sigma = (const float*)d_in[2];
  const float* rho   = (const float*)d_in[3];
  float*       out   = (float*)d_out;

  _Float16* Bpack = (_Float16*)d_ws;
  float*    bias  = (float*)((char*)d_ws + (size_t)(2 * NTILE * 32 * 16) * sizeof(_Float16));

  fuzzy_prep_kernel<<<1, 256, 0, stream>>>(mu, sigma, rho, Bpack, bias);

  const int n         = in_sizes[0] / A_DIM;
  const int row_tiles = n / 16;
  const int blocks    = 256;
  const int threads   = 256;                   // 8 waves per WGP-resident block
  const int total_waves = blocks * threads / 32;

  fuzzy_main_kernel<<<blocks, threads, 0, stream>>>(x, Bpack, bias, out,
                                                    row_tiles, total_waves);
}
